// LatentKVAttention_22153441313015
// MI455X (gfx1250) — compile-verified, matrix operand order checked by reading, not run on hardware
//
#include <hip/hip_runtime.h>

#define N_HEADS 16
#define N_KV    4
#define D_HEAD  128
#define R_K     256
#define R_V     384
#define BATCH   4
#define SEQ     2048
#define MTOT    (BATCH * SEQ)      // 8192
#define DMODEL  2048

typedef __attribute__((ext_vector_type(8)))  __bf16 bf16x8;
typedef __attribute__((ext_vector_type(16))) __bf16 bf16x16;
typedef __attribute__((ext_vector_type(8)))  float  f32x8;

// ---------------------------------------------------------------------------
// CDNA5 async global->LDS copy (ASYNCcnt-tracked, bypasses VGPRs).
// Generic pointers to __shared__ carry the LDS byte offset in their low 32
// bits (aperture rules, CDNA5 ISA), which is exactly the dsaddr
// the instruction wants.
// ---------------------------------------------------------------------------
__device__ __forceinline__ void async_copy_b128(void* lds, const void* gptr) {
  asm volatile("global_load_async_to_lds_b128 %0, %1, off"
               :: "v"((unsigned int)(uintptr_t)lds), "v"(gptr)
               : "memory");
}
__device__ __forceinline__ void wait_async0() {
  asm volatile("s_wait_asynccnt 0x0" ::: "memory");
}

// ---------------------------------------------------------------------------
// WMMA helpers (CDNA5 16x16x32 bf16 -> f32)
// ---------------------------------------------------------------------------
__device__ __forceinline__ f32x8 wmma_bf16(bf16x16 a, bf16x16 b, f32x8 c) {
  return __builtin_amdgcn_wmma_f32_16x16x32_bf16(false, a, false, b, (short)0, c,
                                                 false, false);
}

// 16x32 bf16 fragment in the CDNA5 A/B per-lane layout from a row-major tile
// (rows = M for A; rows = N for B, i.e. W stored [N,K] row-major).
// lane 0-15: row=lane,    K = {0..7, 16..23}
// lane16-31: row=lane-16, K = {8..15, 24..31}
__device__ __forceinline__ bf16x16 load_frag(const __bf16* p0, int ld, int lane) {
  const int r  = lane & 15;
  const int kk = (lane >> 4) << 3;  // 0 or 8
  const __bf16* p = p0 + (size_t)r * ld + kk;
  union { bf16x16 v; bf16x8 h[2]; } u;
  u.h[0] = *(const bf16x8*)(p);
  u.h[1] = *(const bf16x8*)(p + 16);
  return u.v;
}

// ---------------------------------------------------------------------------
// fp32 -> bf16 conversion
// ---------------------------------------------------------------------------
__global__ void k_cvt_bf16(const float* __restrict__ in, __bf16* __restrict__ out,
                           int n) {
  int i = blockIdx.x * 256 + threadIdx.x;
  if (i < n) out[i] = (__bf16)in[i];
}

// ---------------------------------------------------------------------------
// C[M,N] = A[M,K](bf16,rm) @ B[N,K](bf16,rm)^T, fp32 accumulate.
// Tile 128x128x32; 128 threads = 4 waves in 2x2; wave tile 64x64
// (16 accumulators). Double-buffered LDS filled with async-to-LDS copies.
// ---------------------------------------------------------------------------
template <typename TOut>
__global__ __launch_bounds__(128) void k_gemm_bf16(const __bf16* __restrict__ A,
                                                   const __bf16* __restrict__ B,
                                                   TOut* __restrict__ C, int M,
                                                   int N, int K) {
  __shared__ __align__(16) __bf16 As[2][128][40];  // +8 pad keeps 16B alignment
  __shared__ __align__(16) __bf16 Bs[2][128][40];

  const int tid  = threadIdx.x;
  const int lane = tid & 31;
  const int wave = tid >> 5;         // 0..3
  const int wm   = wave >> 1;        // 0..1
  const int wn   = wave & 1;         // 0..1
  const int m0   = blockIdx.y * 128;
  const int n0   = blockIdx.x * 128;
  const int lh   = lane >> 4;
  const int col  = lane & 15;

  f32x8 acc[4][4] = {};

  // stage one 128x32 slab of A and B: 512 16B chunks each, 4/thread each
  auto issue = [&](int st, int k0) {
#pragma unroll
    for (int i = 0; i < 4; ++i) {
      int c  = tid + i * 128;
      int r  = c >> 2;
      int cc = (c & 3) << 3;
      async_copy_b128(&As[st][r][cc], A + (size_t)(m0 + r) * K + k0 + cc);
      async_copy_b128(&Bs[st][r][cc], B + (size_t)(n0 + r) * K + k0 + cc);
    }
  };

  const int nk = K >> 5;
  issue(0, 0);
  wait_async0();
  __syncthreads();

  for (int kb = 0; kb < nk; ++kb) {
    const int cur = kb & 1;
    if (kb + 1 < nk) issue(cur ^ 1, (kb + 1) << 5);

    bf16x16 bfr[4];
#pragma unroll
    for (int j = 0; j < 4; ++j)
      bfr[j] = load_frag(&Bs[cur][wn * 64 + j * 16][0], 40, lane);
#pragma unroll
    for (int i = 0; i < 4; ++i) {
      bf16x16 af = load_frag(&As[cur][wm * 64 + i * 16][0], 40, lane);
#pragma unroll
      for (int j = 0; j < 4; ++j) acc[i][j] = wmma_bf16(af, bfr[j], acc[i][j]);
    }

    if (kb + 1 < nk) wait_async0();
    __syncthreads();
  }

#pragma unroll
  for (int i = 0; i < 4; ++i) {
#pragma unroll
    for (int j = 0; j < 4; ++j) {
      int rbase = m0 + wm * 64 + i * 16 + lh * 8;
      int cbase = n0 + wn * 64 + j * 16 + col;
#pragma unroll
      for (int v = 0; v < 8; ++v)
        C[(size_t)(rbase + v) * N + cbase] = (TOut)acc[i][j][v];
    }
  }
}

// ---------------------------------------------------------------------------
// RoPE for Q (folds in 1/sqrt(d) attention scale): q32[MTOT,2048] ->
// qb[B,H,S,128] bf16
// ---------------------------------------------------------------------------
__global__ void k_rope_q(const float* __restrict__ q32, __bf16* __restrict__ qb) {
  int idx = blockIdx.x * 256 + threadIdx.x;
  if (idx >= MTOT * N_HEADS * 64) return;
  int j   = idx & 63;
  int h   = (idx >> 6) & (N_HEADS - 1);
  int row = idx >> 10;
  int b   = row >> 11;           // / SEQ
  int s   = row & (SEQ - 1);
  const float scale = 0.08838834764831845f;  // 1/sqrt(128)
  float inv = __powf(10000.0f, -(float)(2 * j) * (1.0f / 128.0f));
  float sn, cs;
  __sincosf((float)s * inv, &sn, &cs);
  const float* src = q32 + (size_t)row * DMODEL + h * D_HEAD;
  float x1 = src[j], x2 = src[j + 64];
  __bf16* dst = qb + (((size_t)(b * N_HEADS + h) * SEQ + s) * D_HEAD);
  dst[j]      = (__bf16)((x1 * cs - x2 * sn) * scale);
  dst[j + 64] = (__bf16)((x2 * cs + x1 * sn) * scale);
}

// RoPE + anchor for K: k32[MTOT,512] -> kb[B,4,S,128]
__global__ void k_rope_k(const float* __restrict__ k32, const float* __restrict__ anc,
                         __bf16* __restrict__ kb) {
  int idx = blockIdx.x * 256 + threadIdx.x;
  if (idx >= MTOT * N_KV * 64) return;
  int j   = idx & 63;
  int h   = (idx >> 6) & (N_KV - 1);
  int row = idx >> 8;
  int b   = row >> 11;
  int s   = row & (SEQ - 1);
  float inv = __powf(10000.0f, -(float)(2 * j) * (1.0f / 128.0f));
  float sn, cs;
  __sincosf((float)s * inv, &sn, &cs);
  const float* src = k32 + (size_t)row * (N_KV * D_HEAD) + h * D_HEAD;
  float x1 = src[j] + anc[h * D_HEAD + j];
  float x2 = src[j + 64] + anc[h * D_HEAD + j + 64];
  __bf16* dst = kb + (((size_t)(b * N_KV + h) * SEQ + s) * D_HEAD);
  dst[j]      = (__bf16)(x1 * cs - x2 * sn);
  dst[j + 64] = (__bf16)(x2 * cs + x1 * sn);
}

// V + anchor: v32[MTOT,512] -> vb[B,4,S,128]
__global__ void k_pack_v(const float* __restrict__ v32, const float* __restrict__ anc,
                         __bf16* __restrict__ vb) {
  int idx = blockIdx.x * 256 + threadIdx.x;
  if (idx >= MTOT * N_KV * D_HEAD) return;
  int i   = idx & 127;
  int h   = (idx >> 7) & (N_KV - 1);
  int row = idx >> 9;
  int b   = row >> 11;
  int s   = row & (SEQ - 1);
  vb[(((size_t)(b * N_KV + h) * SEQ + s) * D_HEAD) + i] =
      (__bf16)(v32[(size_t)row * (N_KV * D_HEAD) + h * D_HEAD + i] + anc[h * D_HEAD + i]);
}

// ---------------------------------------------------------------------------
// Causal flash attention, bf16 WMMA, fp32 online softmax.
// Grid (S/64, H, B); 128 threads = 4 waves; wave owns 16 query rows.
// Q pre-scaled by 1/sqrt(d) in k_rope_q. K block staged via async-to-LDS.
// ---------------------------------------------------------------------------
__global__ __launch_bounds__(128) void k_attn(const __bf16* __restrict__ qb,
                                              const __bf16* __restrict__ kb,
                                              const __bf16* __restrict__ vb,
                                              float* __restrict__ o32) {
  __shared__ __align__(16) __bf16 Ks[64][136];   // keys x d
  __shared__ __align__(16) __bf16 Vt[128][72];   // d x keys (transposed)
  __shared__ __align__(16) __bf16 Ps[4][16][72]; // per-wave P staging

  const int tid  = threadIdx.x;
  const int lane = tid & 31;
  const int wave = tid >> 5;
  const int qblk = blockIdx.x;
  const int h    = blockIdx.y;
  const int b    = blockIdx.z;
  const int hkv  = h >> 2;  // groups = H / Hkv = 4
  const int q0   = qblk * 64 + wave * 16;
  const int lh   = lane >> 4;
  const int col  = lane & 15;

  const __bf16* qbase = qb + ((size_t)(b * N_HEADS + h) * SEQ) * D_HEAD;
  const __bf16* kbase = kb + ((size_t)(b * N_KV + hkv) * SEQ) * D_HEAD;
  const __bf16* vbase = vb + ((size_t)(b * N_KV + hkv) * SEQ) * D_HEAD;

  bf16x16 qf[4];
#pragma unroll
  for (int kt = 0; kt < 4; ++kt)
    qf[kt] = load_frag(qbase + (size_t)q0 * D_HEAD + kt * 32, D_HEAD, lane);

  f32x8 oacc[8] = {};
  float mrun[8], lrun[8];
#pragma unroll
  for (int v = 0; v < 8; ++v) { mrun[v] = -3.0e38f; lrun[v] = 0.0f; }

  for (int kv0 = 0; kv0 <= qblk * 64; kv0 += 64) {
    // stage K async (64x128 = 1024 chunks, 8/thread); V transposed via regs
#pragma unroll
    for (int i = 0; i < 8; ++i) {
      int c  = tid + i * 128;
      int r  = c >> 4;
      int cc = (c & 15) << 3;
      async_copy_b128(&Ks[r][cc], kbase + (size_t)(kv0 + r) * D_HEAD + cc);
      bf16x8 vv = *(const bf16x8*)(vbase + (size_t)(kv0 + r) * D_HEAD + cc);
#pragma unroll
      for (int e = 0; e < 8; ++e) Vt[cc + e][r] = vv[e];
    }
    wait_async0();
    __syncthreads();

    // scores: 4 tiles of 16 keys, each = 4 chained WMMAs over d
    f32x8 sac[4];
#pragma unroll
    for (int nt = 0; nt < 4; ++nt) {
      f32x8 s = {};
#pragma unroll
      for (int kt = 0; kt < 4; ++kt)
        s = wmma_bf16(qf[kt], load_frag(&Ks[nt * 16][kt * 32], 136, lane), s);
      sac[nt] = s;
    }

    // causal mask + online softmax (C layout: row = v + 8*lh, col = lane&15)
#pragma unroll
    for (int v = 0; v < 8; ++v) {
      int qrow = q0 + v + lh * 8;
      float mx = -3.0e38f;
#pragma unroll
      for (int nt = 0; nt < 4; ++nt) {
        int key  = kv0 + nt * 16 + col;
        float sv = sac[nt][v];
        sv       = (key <= qrow) ? sv : -3.0e38f;
        sac[nt][v] = sv;
        mx = fmaxf(mx, sv);
      }
#pragma unroll
      for (int mk = 1; mk < 16; mk <<= 1) mx = fmaxf(mx, __shfl_xor(mx, mk, 32));
      float nm    = fmaxf(mrun[v], mx);
      float alpha = __expf(mrun[v] - nm);
      mrun[v]     = nm;
      float rs = 0.0f;
#pragma unroll
      for (int nt = 0; nt < 4; ++nt) {
        float p = __expf(sac[nt][v] - nm);
        sac[nt][v] = p;
        rs += p;
      }
#pragma unroll
      for (int mk = 1; mk < 16; mk <<= 1) rs += __shfl_xor(rs, mk, 32);
      lrun[v] = lrun[v] * alpha + rs;
#pragma unroll
      for (int n = 0; n < 8; ++n) oacc[n][v] *= alpha;
    }

    // restage P as bf16 A-fragments (same-wave LDS ordering, no barrier)
#pragma unroll
    for (int nt = 0; nt < 4; ++nt)
#pragma unroll
      for (int v = 0; v < 8; ++v)
        Ps[wave][v + lh * 8][nt * 16 + col] = (__bf16)sac[nt][v];

    // out += P @ V : K-dim = 64 keys (2 chunks), N = 8 tiles of d
#pragma unroll
    for (int kt = 0; kt < 2; ++kt) {
      bf16x16 pf = load_frag(&Ps[wave][0][kt * 32], 72, lane);
#pragma unroll
      for (int n = 0; n < 8; ++n)
        oacc[n] = wmma_bf16(pf, load_frag(&Vt[n * 16][kt * 32], 72, lane), oacc[n]);
    }
    __syncthreads();
  }

  float* obase = o32 + (((size_t)(b * N_HEADS + h) * SEQ + q0) * D_HEAD);
#pragma unroll
  for (int v = 0; v < 8; ++v) {
    float invl = 1.0f / lrun[v];
    int r = v + lh * 8;
#pragma unroll
    for (int n = 0; n < 8; ++n)
      obase[(size_t)r * D_HEAD + n * 16 + col] = oacc[n][v] * invl;
  }
}

// o32[B,H,S,128] -> ob[MTOT, 2048] bf16 (heads interleaved back)
__global__ void k_reshape_o(const float* __restrict__ o32, __bf16* __restrict__ ob) {
  int idx = blockIdx.x * 256 + threadIdx.x;
  if (idx >= MTOT * DMODEL) return;
  int c   = idx & (DMODEL - 1);
  int row = idx >> 11;
  int h   = c >> 7;
  int i   = c & 127;
  int b   = row >> 11;
  int s   = row & (SEQ - 1);
  ob[idx] = (__bf16)o32[(((size_t)(b * N_HEADS + h) * SEQ + s) * D_HEAD) + i];
}

// ---------------------------------------------------------------------------
extern "C" void kernel_launch(void* const* d_in, const int* in_sizes, int n_in,
                              void* d_out, int out_size, void* d_ws, size_t ws_size,
                              hipStream_t stream) {
  (void)in_sizes; (void)n_in; (void)out_size; (void)ws_size;
  const float* hs  = (const float*)d_in[0];
  const float* Wq  = (const float*)d_in[1];
  const float* Wkd = (const float*)d_in[2];
  const float* Wku = (const float*)d_in[3];
  const float* Wvd = (const float*)d_in[4];
  const float* Wvu = (const float*)d_in[5];
  const float* ka  = (const float*)d_in[6];
  const float* va  = (const float*)d_in[7];
  const float* Wo  = (const float*)d_in[8];
  float* out = (float*)d_out;

  char* ws = (char*)d_ws;
  size_t off = 0;
  auto alloc = [&](size_t bytes) -> void* {
    void* p = ws + off;
    off = (off + bytes + 255) & ~(size_t)255;
    return p;
  };

  __bf16* hb   = (__bf16*)alloc((size_t)MTOT * DMODEL * 2);
  __bf16* wqb  = (__bf16*)alloc((size_t)DMODEL * DMODEL * 2);
  __bf16* wkdb = (__bf16*)alloc((size_t)R_K * DMODEL * 2);
  __bf16* wkub = (__bf16*)alloc((size_t)(N_KV * D_HEAD) * R_K * 2);
  __bf16* wvdb = (__bf16*)alloc((size_t)R_V * DMODEL * 2);
  __bf16* wvub = (__bf16*)alloc((size_t)(N_KV * D_HEAD) * R_V * 2);
  __bf16* wob  = (__bf16*)alloc((size_t)DMODEL * DMODEL * 2);
  float*  q32  = (float*)alloc((size_t)MTOT * DMODEL * 4);
  __bf16* klb  = (__bf16*)alloc((size_t)MTOT * R_K * 2);
  float*  k32  = (float*)alloc((size_t)MTOT * (N_KV * D_HEAD) * 4);
  __bf16* vlb  = (__bf16*)alloc((size_t)MTOT * R_V * 2);
  float*  v32  = (float*)alloc((size_t)MTOT * (N_KV * D_HEAD) * 4);
  __bf16* qbh  = (__bf16*)alloc((size_t)MTOT * DMODEL * 2);
  __bf16* kbh  = (__bf16*)alloc((size_t)MTOT * (N_KV * D_HEAD) * 2);
  __bf16* vbh  = (__bf16*)alloc((size_t)MTOT * (N_KV * D_HEAD) * 2);
  float*  o32  = q32;   // alias: q32 dead after k_rope_q
  __bf16* ob   = hb;    // alias: hb dead after V-down GEMM

  auto cvt = [&](const float* src, __bf16* dst, int n) {
    k_cvt_bf16<<<(n + 255) / 256, 256, 0, stream>>>(src, dst, n);
  };
  cvt(hs,  hb,   MTOT * DMODEL);
  cvt(Wq,  wqb,  DMODEL * DMODEL);
  cvt(Wkd, wkdb, R_K * DMODEL);
  cvt(Wku, wkub, N_KV * D_HEAD * R_K);
  cvt(Wvd, wvdb, R_V * DMODEL);
  cvt(Wvu, wvub, N_KV * D_HEAD * R_V);
  cvt(Wo,  wob,  DMODEL * DMODEL);

  // projections
  k_gemm_bf16<float><<<dim3(DMODEL / 128, MTOT / 128), 128, 0, stream>>>(
      hb, wqb, q32, MTOT, DMODEL, DMODEL);
  k_gemm_bf16<__bf16><<<dim3(R_K / 128, MTOT / 128), 128, 0, stream>>>(
      hb, wkdb, klb, MTOT, R_K, DMODEL);
  k_gemm_bf16<float><<<dim3((N_KV * D_HEAD) / 128, MTOT / 128), 128, 0, stream>>>(
      klb, wkub, k32, MTOT, N_KV * D_HEAD, R_K);
  k_gemm_bf16<__bf16><<<dim3(R_V / 128, MTOT / 128), 128, 0, stream>>>(
      hb, wvdb, vlb, MTOT, R_V, DMODEL);
  k_gemm_bf16<float><<<dim3((N_KV * D_HEAD) / 128, MTOT / 128), 128, 0, stream>>>(
      vlb, wvub, v32, MTOT, N_KV * D_HEAD, R_V);

  // rope / anchor / layout
  {
    int n = MTOT * N_HEADS * 64;
    k_rope_q<<<(n + 255) / 256, 256, 0, stream>>>(q32, qbh);
  }
  {
    int n = MTOT * N_KV * 64;
    k_rope_k<<<(n + 255) / 256, 256, 0, stream>>>(k32, ka, kbh);
  }
  {
    int n = MTOT * N_KV * D_HEAD;
    k_pack_v<<<(n + 255) / 256, 256, 0, stream>>>(v32, va, vbh);
  }

  // attention
  k_attn<<<dim3(SEQ / 64, N_HEADS, BATCH), 128, 0, stream>>>(qbh, kbh, vbh, o32);

  // output projection
  {
    int n = MTOT * DMODEL;
    k_reshape_o<<<(n + 255) / 256, 256, 0, stream>>>(o32, ob);
  }
  k_gemm_bf16<float><<<dim3(DMODEL / 128, MTOT / 128), 128, 0, stream>>>(
      ob, wob, out, MTOT, DMODEL, DMODEL);
}
